// DecoderRNNmodel_37658273251908
// MI455X (gfx1250) — compile-verified, matrix operand order checked by reading, not run on hardware
//
#include <hip/hip_runtime.h>

// ---------------- problem constants ----------------
#define T_STEPS 2048
#define BATCH   64
#define HID     1024
#define NWG     64          // workgroups; each owns SLICE hidden units
#define SLICE   16
#define KPAD    1032        // 1024 + 8 bf16 pad -> 2064B row stride (conflict-free b128)

typedef __attribute__((ext_vector_type(16))) __bf16 v16bf;
typedef __attribute__((ext_vector_type(8)))  __bf16 v8bf;
typedef __attribute__((ext_vector_type(8)))  float  v8f;

// LDS layout (bytes)
#define WLDS_BYTES   (64 * KPAD * 2)              // 132096 : W_hh slice, bf16 [gn=64][KPAD]
#define HLDS_BYTES   (64 * KPAD * 2)              // 132096 : h_prev stage, bf16 [b=64][KPAD]
#define CTRL_BYTES   (16384 + 4096 + 1024 + 1024 + 256 + 256 + 1024)
#define SMEM_BYTES   (WLDS_BYTES + HLDS_BYTES + CTRL_BYTES)   // 288256 < 320KB WGP LDS

__device__ __forceinline__ float sigmoidf_(float x) { return 1.0f / (1.0f + __expf(-x)); }

__device__ __forceinline__ void grid_sync(unsigned* cnt, unsigned* gen, unsigned nwg) {
    __builtin_amdgcn_fence(__ATOMIC_RELEASE, "agent");
    __syncthreads();
    if (threadIdx.x == 0) {
        unsigned g = __hip_atomic_load(gen, __ATOMIC_RELAXED, __HIP_MEMORY_SCOPE_AGENT);
        if (__hip_atomic_fetch_add(cnt, 1u, __ATOMIC_ACQ_REL, __HIP_MEMORY_SCOPE_AGENT) == nwg - 1u) {
            __hip_atomic_store(cnt, 0u, __ATOMIC_RELAXED, __HIP_MEMORY_SCOPE_AGENT);
            __hip_atomic_fetch_add(gen, 1u, __ATOMIC_ACQ_REL, __HIP_MEMORY_SCOPE_AGENT);
        } else {
            while (__hip_atomic_load(gen, __ATOMIC_ACQUIRE, __HIP_MEMORY_SCOPE_AGENT) == g) {
                __builtin_amdgcn_s_sleep(8);
            }
        }
    }
    __syncthreads();
    __builtin_amdgcn_fence(__ATOMIC_ACQUIRE, "agent");
}

__global__ __launch_bounds__(256, 1)
void lstm_persistent(const float* __restrict__ x,     // [T,B,4]
                     const float* __restrict__ h0,    // [B,H]
                     const float* __restrict__ c0,    // [B,H]
                     const float* __restrict__ Wih,   // [4H,4]
                     const float* __restrict__ Whh,   // [4H,H]
                     const float* __restrict__ bih,   // [4H]
                     const float* __restrict__ bhh,   // [4H]
                     const float* __restrict__ Wlin,  // [4,H]
                     const float* __restrict__ blin,  // [4]
                     float* __restrict__ out,         // [T*B,4]
                     __bf16* __restrict__ hbuf,       // ws: [2][B][H] bf16
                     float* __restrict__ partials,    // ws: [2][NWG][B][4]
                     unsigned* __restrict__ bar)      // ws: {cnt, gen}
{
    extern __shared__ char smem[];
    __bf16* Wlds  = (__bf16*)smem;                             // [64][KPAD] bf16
    __bf16* Hlds  = (__bf16*)(smem + WLDS_BYTES);              // [64][KPAD] bf16
    float*  gates = (float*)(smem + WLDS_BYTES + HLDS_BYTES);  // [4][64][16]
    float*  cst   = gates + 4 * BATCH * SLICE;                 // [64][16]
    float*  xt    = cst   + BATCH * SLICE;                     // [64][4]
    float*  wihL  = xt    + BATCH * 4;                         // [64][4]  (gn-major)
    float*  biasL = wihL  + 64 * 4;                            // [64]
    float*  wlinL = biasL + 64;                                // [4][16]
    float*  plin  = wlinL + 64;                                // [64][4]

    const int tid   = threadIdx.x;
    const int wg    = blockIdx.x;          // 0..63
    const int n0    = wg * SLICE;
    const int wave  = tid >> 5;
    const int lane  = tid & 31;
    const int nl    = lane & 15;
    const int kh    = lane >> 4;           // K-half select for 16-bit operand layout
    const int mtile = wave & 3;            // which 16 batch rows
    const int gp    = wave >> 2;           // gate pair {0,1} or {2,3}
    const int g0    = gp * 2, g1 = gp * 2 + 1;

    // ---- one-time: W_hh slice -> LDS as bf16 (coalesced, row-padded) ----
    for (int idx = tid; idx < 64 * HID; idx += 256) {
        int gn = idx >> 10;                // g*16+n
        int k  = idx & (HID - 1);
        int g  = gn >> 4, n = gn & 15;
        Wlds[gn * KPAD + k] = (__bf16)Whh[(size_t)(g * HID + n0 + n) * HID + k];
    }
    if (tid < 64) {
        int g = tid >> 4, n = tid & 15;
        int r = g * HID + n0 + n;
        biasL[tid] = bih[r] + bhh[r];
        #pragma unroll
        for (int ii = 0; ii < 4; ++ii) wihL[tid * 4 + ii] = Wih[r * 4 + ii];
        wlinL[tid] = Wlin[(tid >> 4) * HID + n0 + (tid & 15)];
    }
    // ---- init: c0 slice -> LDS, h0 slice -> hbuf[0] (bf16) ----
    for (int idx = tid; idx < BATCH * SLICE; idx += 256) {
        int b = idx >> 4, n = idx & 15;
        cst[idx] = c0[b * HID + n0 + n];
        hbuf[b * HID + n0 + n] = (__bf16)h0[b * HID + n0 + n];
    }
    grid_sync(bar, bar + 1, NWG);

    // ---- recurrence ----
    for (int t = 0; t < T_STEPS; ++t) {
        const int par = t & 1;
        const __bf16* __restrict__ hprev = hbuf + (size_t)par * BATCH * HID;

        // async bulk copy of h_prev (128KB contiguous bf16) into padded LDS stage.
        // 8192 16-byte chunks; chunk q -> row q>>7, chunk-in-row q&127.
        // VDST operand is the LDS byte offset (no static LDS, dynamic base == 0).
        {
            #pragma unroll 8
            for (int i = 0; i < 32; ++i) {
                int q   = tid + i * 256;
                int row = q >> 7;
                int c   = q & 127;
                unsigned laddr = (unsigned)(WLDS_BYTES + row * (KPAD * 2) + c * 16);
                const __bf16* g = hprev + (size_t)q * 8;
                asm volatile("global_load_async_to_lds_b128 %0, %1, off"
                             :: "v"(laddr), "v"(g) : "memory");
            }
            asm volatile("s_wait_asynccnt 0x0" ::: "memory");
        }

        // WG0 finalizes output head for step t-1 (overlaps with t's GEMM elsewhere)
        if (wg == 0 && t > 0) {
            int pprev = (t - 1) & 1;
            int b = tid >> 2, j = tid & 3;
            float s = blin[j];
            const float* pp = partials + (size_t)pprev * NWG * BATCH * 4 + b * 4 + j;
            #pragma unroll 8
            for (int w2 = 0; w2 < NWG; ++w2) s += pp[(size_t)w2 * BATCH * 4];
            out[((size_t)(t - 1) * BATCH + b) * 4 + j] = (j < 2) ? tanhf(s) : sigmoidf_(s);
        }

        // stage x[t] (64x4 f32), zero linear partials
        if (tid < BATCH * 4) {
            xt[tid]   = x[(size_t)t * BATCH * 4 + tid];
            plin[tid] = 0.0f;
        }
        if (tid == 0 && t + 1 < T_STEPS)
            __builtin_prefetch(&x[(size_t)(t + 1) * BATCH * 4], 0, 1);  // global_prefetch_b8
        __syncthreads();   // async h stage + x[t] visible to all waves

        // accumulators seeded with x-gates + bias (C layout: vgpr j -> M=j+8*kh, N=nl)
        v8f acc0 = {}, acc1 = {};
        #pragma unroll
        for (int j = 0; j < 8; ++j) {
            int b = mtile * 16 + j + 8 * kh;
            float xv0 = biasL[g0 * 16 + nl];
            float xv1 = biasL[g1 * 16 + nl];
            #pragma unroll
            for (int ii = 0; ii < 4; ++ii) {
                float xb = xt[b * 4 + ii];
                xv0 = fmaf(xb, wihL[(g0 * 16 + nl) * 4 + ii], xv0);
                xv1 = fmaf(xb, wihL[(g1 * 16 + nl) * 4 + ii], xv1);
            }
            acc0[j] = xv0;
            acc1[j] = xv1;
        }

        // K loop: A = h rows from LDS stage, B = W slice from LDS (all ds_load_b128)
        const __bf16* __restrict__ arow = Hlds + (mtile * 16 + nl) * KPAD + kh * 8;
        const __bf16* __restrict__ b0p  = Wlds + (g0 * 16 + nl) * KPAD + kh * 8;
        const __bf16* __restrict__ b1p  = Wlds + (g1 * 16 + nl) * KPAD + kh * 8;
        #pragma unroll 4
        for (int kk = 0; kk < HID / 32; ++kk) {
            const int ko = kk * 32;
            v8bf alo = *(const v8bf*)(arow + ko);
            v8bf ahi = *(const v8bf*)(arow + ko + 16);
            v16bf a = __builtin_shufflevector(alo, ahi, 0,1,2,3,4,5,6,7,8,9,10,11,12,13,14,15);
            v8bf b0lo = *(const v8bf*)(b0p + ko);
            v8bf b0hi = *(const v8bf*)(b0p + ko + 16);
            v16bf bm0 = __builtin_shufflevector(b0lo, b0hi, 0,1,2,3,4,5,6,7,8,9,10,11,12,13,14,15);
            acc0 = __builtin_amdgcn_wmma_f32_16x16x32_bf16(false, a, false, bm0, (short)0, acc0, false, false);
            v8bf b1lo = *(const v8bf*)(b1p + ko);
            v8bf b1hi = *(const v8bf*)(b1p + ko + 16);
            v16bf bm1 = __builtin_shufflevector(b1lo, b1hi, 0,1,2,3,4,5,6,7,8,9,10,11,12,13,14,15);
            acc1 = __builtin_amdgcn_wmma_f32_16x16x32_bf16(false, a, false, bm1, (short)0, acc1, false, false);
        }

        // spill gate tiles to LDS for cross-wave gate combination
        #pragma unroll
        for (int j = 0; j < 8; ++j) {
            int b = mtile * 16 + j + 8 * kh;
            gates[(g0 * BATCH + b) * 16 + nl] = acc0[j];
            gates[(g1 * BATCH + b) * 16 + nl] = acc1[j];
        }
        __syncthreads();

        // LSTM cell update (local to this WG's hidden slice) + output-head partials
        const int nxt = par ^ 1;
        __bf16* __restrict__ hnext = hbuf + (size_t)nxt * BATCH * HID;
        #pragma unroll
        for (int p = 0; p < 4; ++p) {
            int idx = tid + p * 256;          // 0..1023 -> (b,n)
            int b = idx >> 4, n = idx & 15;
            float gi = gates[(0 * BATCH + b) * 16 + n];
            float gf = gates[(1 * BATCH + b) * 16 + n];
            float gc = gates[(2 * BATCH + b) * 16 + n];
            float go = gates[(3 * BATCH + b) * 16 + n];
            float c  = cst[idx];
            float cn = sigmoidf_(gf) * c + sigmoidf_(gi) * tanhf(gc);
            float h  = sigmoidf_(go) * tanhf(cn);
            cst[idx] = cn;
            hnext[(size_t)b * HID + n0 + n] = (__bf16)h;
            #pragma unroll
            for (int j = 0; j < 4; ++j)
                atomicAdd(&plin[b * 4 + j], h * wlinL[j * 16 + n]);
        }
        __syncthreads();
        if (tid < BATCH * 4)
            partials[(size_t)par * NWG * BATCH * 4 + (size_t)wg * BATCH * 4 + tid] = plin[tid];

        grid_sync(bar, bar + 1, NWG);
    }

    // output head for the final step
    if (wg == 0) {
        int pprev = (T_STEPS - 1) & 1;
        int b = tid >> 2, j = tid & 3;
        float s = blin[j];
        const float* pp = partials + (size_t)pprev * NWG * BATCH * 4 + b * 4 + j;
        #pragma unroll 8
        for (int w2 = 0; w2 < NWG; ++w2) s += pp[(size_t)w2 * BATCH * 4];
        out[((size_t)(T_STEPS - 1) * BATCH + b) * 4 + j] = (j < 2) ? tanhf(s) : sigmoidf_(s);
    }
}

extern "C" void kernel_launch(void* const* d_in, const int* in_sizes, int n_in,
                              void* d_out, int out_size, void* d_ws, size_t ws_size,
                              hipStream_t stream) {
    (void)in_sizes; (void)n_in; (void)out_size; (void)ws_size;
    const float* x    = (const float*)d_in[0];
    const float* h0   = (const float*)d_in[1];
    const float* c0   = (const float*)d_in[2];
    const float* Wih  = (const float*)d_in[3];
    const float* Whh  = (const float*)d_in[4];
    const float* bih  = (const float*)d_in[5];
    const float* bhh  = (const float*)d_in[6];
    const float* Wlin = (const float*)d_in[7];
    const float* blin = (const float*)d_in[8];
    float* out = (float*)d_out;

    unsigned char* ws = (unsigned char*)d_ws;
    unsigned* bar    = (unsigned*)ws;                                   // cnt, gen
    __bf16*  hbuf    = (__bf16*)(ws + 256);                             // 2*64*1024 bf16 = 256KB
    float*   parts   = (float*)(ws + 256 + (size_t)2 * BATCH * HID * 2);// 2*64*64*4 f32 = 128KB

    hipMemsetAsync(bar, 0, 256, stream);   // barrier state must start at 0 (ws is poisoned)

    lstm_persistent<<<dim3(NWG), dim3(256), SMEM_BYTES, stream>>>(
        x, h0, c0, Wih, Whh, bih, bhh, Wlin, blin, out, hbuf, parts, bar);
}
